// CompressionGPT_13073880449871
// MI455X (gfx1250) — matrix operand order repaired and checked by reading, not compiled, not run on hardware
//
#include <hip/hip_runtime.h>
#include <math.h>

typedef __attribute__((ext_vector_type(16))) __bf16 v16bf;
typedef __attribute__((ext_vector_type(8)))  float  v8f;
typedef __attribute__((ext_vector_type(4)))  unsigned int u32x4;
typedef __attribute__((ext_vector_type(4)))  int i32x4;
typedef __attribute__((ext_vector_type(8)))  int i32x8;

constexpr int Bsz  = 4;
constexpr int T    = 1024;
constexpr int E    = 512;
constexpr int H    = 8;
constexpr int Lnum = 4;
constexpr int Vv   = 32000;
constexpr int HD   = E / H;     // 64
constexpr int M    = Bsz * T;   // 4096
constexpr int RS   = 3 * E;     // qkv row stride (elements)

union ABFrag { v16bf v; uint4 u[2]; unsigned short s[16]; };

static __device__ __forceinline__ unsigned short f2bf(float f) {
  union { float f; unsigned int u; } cv; cv.f = f;
  unsigned int u = cv.u;
  u += 0x7FFFu + ((u >> 16) & 1u);   // round-to-nearest-even
  return (unsigned short)(u >> 16);
}

static __device__ __forceinline__ float gelu_exact(float v) {
  return 0.5f * v * (1.0f + erff(v * 0.70710678118654752f));
}

// LDS byte address = low 32 bits of the generic (flat) address of a __shared__
// object (shared aperture: addr[31:0] is the in-group LDS offset).
static __device__ __forceinline__ unsigned lds_addr_of(const void* p) {
  return (unsigned)(uintptr_t)p;
}

// ---------------------------------------------------------------------------
// Tensor Data Mover: 2D tile load (tile_dim0 = 32 bf16 elems along the
// contiguous axis, tile_dim1 = tile_rows) from a (rows x row_len) bf16 tensor.
// D# built per CDNA5 ISA 8.3/8.4; groups 2/3 zero (2D tensor).
// ---------------------------------------------------------------------------
static __device__ __forceinline__ void
tdm_load_tile(const unsigned short* gsrc, unsigned lds_byte_addr,
              unsigned tile_rows, unsigned row_len_elems) {
  const unsigned long long ga = (unsigned long long)(uintptr_t)gsrc;
  u32x4 g0;
  g0.x = 1u;                                           // count=1, user mode
  g0.y = lds_byte_addr;                                // D#.lds_addr
  g0.z = (unsigned)ga;                                 // global_addr[31:0]
  g0.w = (unsigned)((ga >> 32) & 0x01FFFFFFu) | (2u << 30); // addr[56:32]|type=2
  const unsigned td0 = row_len_elems;                  // tensor_dim0 = K
  const unsigned td1 = 1u << 20;                       // tensor_dim1 (no OOB)
  i32x8 g1;
  g1[0] = (int)(1u << 16);                             // data_size=1 (2B/elem)
  g1[1] = (int)((td0 & 0xFFFFu) << 16);                // tensor_dim0[15:0]
  g1[2] = (int)((td0 >> 16) | ((td1 & 0xFFFFu) << 16));
  g1[3] = (int)((td1 >> 16) | (32u << 16));            // tile_dim0 = 32
  g1[4] = (int)tile_rows;                              // tile_dim1 (dim2=0)
  g1[5] = (int)row_len_elems;                          // dim0_stride[31:0]
  g1[6] = 0;
  g1[7] = 0;
  i32x4 z4 = {0, 0, 0, 0};
#if defined(__clang_major__) && (__clang_major__ >= 23)
  i32x8 z8 = {0, 0, 0, 0, 0, 0, 0, 0};
  __builtin_amdgcn_tensor_load_to_lds(g0, g1, z4, z4, z8, 0);
#else
  __builtin_amdgcn_tensor_load_to_lds(g0, g1, z4, z4, 0);
#endif
}

// ---------------------------------------------------------------------------
// fp32 -> bf16 conversion (weights)
// ---------------------------------------------------------------------------
__global__ void __launch_bounds__(256)
cvt_f32_bf16(const float* __restrict__ src, unsigned short* __restrict__ dst, int n) {
  int i = blockIdx.x * 256 + threadIdx.x;
  if (i < n) dst[i] = f2bf(src[i]);
}

// ---------------------------------------------------------------------------
// Embedding: x = wte[tok] + (t < idxs_dec[b] ? wpe_enc : wpe)[t]
// ---------------------------------------------------------------------------
__global__ void __launch_bounds__(256)
embed_kernel(const float* __restrict__ wte, const float* __restrict__ wpe,
             const float* __restrict__ wpe_enc, const int* __restrict__ tok,
             const int* __restrict__ idxs, float* __restrict__ x) {
  int bt = blockIdx.x;
  int b = bt / T, t = bt % T;
  int tk = tok[bt];
  const float* pe = (t < idxs[b]) ? wpe_enc : wpe;
  for (int i = threadIdx.x; i < E; i += 256)
    x[(size_t)bt * E + i] = wte[(size_t)tk * E + i] + pe[(size_t)t * E + i];
}

// ---------------------------------------------------------------------------
// LayerNorm (one wave per row of E=512), bf16 output for WMMA A operand
// ---------------------------------------------------------------------------
__global__ void __launch_bounds__(256)
ln_kernel(const float* __restrict__ x, const float* __restrict__ g,
          const float* __restrict__ beta, unsigned short* __restrict__ y) {
  int lane = threadIdx.x & 31, wv = threadIdx.x >> 5;
  int row = blockIdx.x * 8 + wv;
  const float* xr = x + (size_t)row * E;
  float s = 0.f, s2 = 0.f;
  for (int i = lane; i < E; i += 32) { float v = xr[i]; s += v; s2 += v * v; }
  for (int off = 1; off < 32; off <<= 1) {
    s  += __shfl_xor(s, off);
    s2 += __shfl_xor(s2, off);
  }
  float mean = s * (1.f / E);
  float var  = s2 * (1.f / E) - mean * mean;
  float rstd = rsqrtf(var + 1e-5f);
  unsigned short* yr = y + (size_t)row * E;
  for (int i = lane; i < E; i += 32)
    yr[i] = f2bf((xr[i] - mean) * rstd * g[i] + beta[i]);
}

// ---------------------------------------------------------------------------
// WMMA GEMM, TDM-fed and double-buffered through LDS.
// out[m,n] = sum_k A[m,k] * W[n,k] (+bias, +resid, act, NaN mask)
// A: (M x K) bf16 row-major, W: (N x K) bf16 row-major.
// Block = 256 thr = 8 waves = 2(M) x 4(N); block tile 32 x 256; K-step 32.
// Wave 0 drives the TDM; tile k+1 DMA overlaps WMMA on tile k.
// ---------------------------------------------------------------------------
template <int ACT, bool RESID, bool OUTF, bool OUTB, bool NANMASK>
__global__ void __launch_bounds__(256)
gemm_wmma(const unsigned short* __restrict__ A,
          const unsigned short* __restrict__ W,
          const float* __restrict__ bias,
          const float* __restrict__ resid,
          float* __restrict__ outF,
          unsigned short* __restrict__ outB,
          const int* __restrict__ idxs_dec,
          int N, int K) {
  __shared__ unsigned short aT[2][32 * 32];    //  2 KB per buffer
  __shared__ unsigned short bT[2][256 * 32];   // 16 KB per buffer

  const int lane   = threadIdx.x & 31;
  const int wv     = threadIdx.x >> 5;
  const int lane16 = lane & 15;
  const bool hi    = lane >= 16;
  const int mg = wv >> 2, ng = wv & 3;
  const int rowBase = blockIdx.y * 32;
  const int colBase = blockIdx.x * 256;
  const bool issuer = (wv == 0);

  v8f acc[4];
#pragma unroll
  for (int i = 0; i < 4; ++i)
#pragma unroll
    for (int e = 0; e < 8; ++e) acc[i][e] = 0.f;

  const int KT = K >> 5;
  if (issuer) {
    tdm_load_tile(A + (size_t)rowBase * K, lds_addr_of(&aT[0][0]), 32, K);
    tdm_load_tile(W + (size_t)colBase * K, lds_addr_of(&bT[0][0]), 256, K);
  }

  for (int kt = 0; kt < KT; ++kt) {
    const int buf = kt & 1;
    if (issuer) __builtin_amdgcn_s_wait_tensorcnt(0);  // tile kt landed
    __syncthreads();                                   // everyone sees it; prev
                                                       // buf reads all retired
    if (issuer && (kt + 1) < KT) {                     // prefetch tile kt+1
      const int k1 = (kt + 1) << 5;
      tdm_load_tile(A + (size_t)rowBase * K + k1,
                    lds_addr_of(&aT[buf ^ 1][0]), 32, K);
      tdm_load_tile(W + (size_t)colBase * K + k1,
                    lds_addr_of(&bT[buf ^ 1][0]), 256, K);
    }

    ABFrag a;
    const unsigned short* ar = &aT[buf][(mg * 16 + lane16) * 32 + (hi ? 8 : 0)];
    a.u[0] = *(const uint4*)ar;
    a.u[1] = *(const uint4*)(ar + 16);
#pragma unroll
    for (int nt = 0; nt < 4; ++nt) {
      const unsigned short* br =
          &bT[buf][(ng * 64 + nt * 16 + lane16) * 32 + (hi ? 8 : 0)];
      ABFrag bf;
      bf.u[0] = *(const uint4*)br;
      bf.u[1] = *(const uint4*)(br + 16);
      acc[nt] = __builtin_amdgcn_wmma_f32_16x16x32_bf16(
          false, a.v, false, bf.v, (short)0, acc[nt], false, false);
    }
  }

  const int rowW = rowBase + mg * 16;
  const int colW = colBase + ng * 64;
#pragma unroll
  for (int nt = 0; nt < 4; ++nt) {
    const int col = colW + nt * 16 + lane16;
    const float bv = bias ? bias[col] : 0.f;
#pragma unroll
    for (int r = 0; r < 8; ++r) {
      const int row = rowW + (hi ? 8 : 0) + r;
      float v = acc[nt][r] + bv;
      if (ACT == 1) v = gelu_exact(v);
      if (RESID) v += resid[(size_t)row * N + col];
      if (NANMASK) {
        int bb = row / T, tt = row % T;
        if (tt < idxs_dec[bb]) v = __builtin_nanf("");
      }
      if (OUTF) outF[(size_t)row * N + col] = v;
      if (OUTB) outB[(size_t)row * N + col] = f2bf(v);
    }
  }
}

// ---------------------------------------------------------------------------
// Fused flash-style attention. qkv: (B*T) x 3E bf16, layout [q|k|v].
// Block = 128 thr = 4 waves; each wave owns 16 query rows of one (b,h).
// V tiles staged to LDS with async global->LDS copies (ASYNCcnt) that overlap
// the S=QK^T WMMAs and the online softmax; P re-laid-out through LDS.
// ---------------------------------------------------------------------------
__global__ void __launch_bounds__(128)
attn_wmma(const unsigned short* __restrict__ qkv,
          const int* __restrict__ idxs_dec,
          unsigned short* __restrict__ attno) {
  __shared__ unsigned short v_lds[4][32 * HD];  // 4 KB / wave
  __shared__ unsigned short p_lds[4][16 * 32];  // 1 KB / wave

  const int lane   = threadIdx.x & 31;
  const int wv     = threadIdx.x >> 5;
  const int lane16 = lane & 15;
  const bool hi    = lane >= 16;
  const int b = blockIdx.z, h = blockIdx.y;
  const int qbase = blockIdx.x * 64 + wv * 16;
  const int d = idxs_dec[b];
  const int dm1 = (d - 1 > 0) ? (d - 1) : 0;
  const float scale = 0.125f;  // 1/sqrt(64)

  // Q A-fragments (K-dim = HD = 64 -> two 16x32 fragments), kept in registers
  ABFrag aq0, aq1;
  {
    const unsigned short* qrow =
        qkv + (size_t)(b * T + qbase + lane16) * RS + h * HD + (hi ? 8 : 0);
    aq0.u[0] = *(const uint4*)(qrow);
    aq0.u[1] = *(const uint4*)(qrow + 16);
    aq1.u[0] = *(const uint4*)(qrow + 32);
    aq1.u[1] = *(const uint4*)(qrow + 48);
  }

  v8f o[4];
  float mrow[8], lrow[8];
#pragma unroll
  for (int i = 0; i < 4; ++i)
#pragma unroll
    for (int e = 0; e < 8; ++e) o[i][e] = 0.f;
#pragma unroll
  for (int r = 0; r < 8; ++r) { mrow[r] = -1e30f; lrow[r] = 0.f; }

  const int jend = qbase + 16;  // causal: every valid key has c <= r
  for (int j0 = 0; j0 < jend; j0 += 32) {
    // ---- async-stage V tile (32 keys x 64) into LDS; overlaps S + softmax ---
    {
      const unsigned short* vrow =
          qkv + (size_t)(b * T + j0 + lane) * RS + 2 * E + h * HD;
      unsigned vdst = lds_addr_of(&v_lds[wv][lane * HD]);
#define ASYNC_V16(OFS)                                                        \
  asm volatile("global_load_async_to_lds_b128 %0, %1, off offset:" #OFS      \
               ::"v"(vdst), "v"(vrow) : "memory")
      ASYNC_V16(0);  ASYNC_V16(16); ASYNC_V16(32); ASYNC_V16(48);
      ASYNC_V16(64); ASYNC_V16(80); ASYNC_V16(96); ASYNC_V16(112);
#undef ASYNC_V16
    }

    // ---- S = Q * K^T for 32 keys: load all 4 K fragments, then 4 WMMAs ----
    ABFrag kb[2][2];
#pragma unroll
    for (int ct = 0; ct < 2; ++ct)
#pragma unroll
      for (int kh = 0; kh < 2; ++kh) {
        const unsigned short* krow =
            qkv + (size_t)(b * T + j0 + ct * 16 + lane16) * RS + E + h * HD +
            kh * 32 + (hi ? 8 : 0);
        kb[ct][kh].u[0] = ((const uint4*)krow)[0];
        kb[ct][kh].u[1] = ((const uint4*)krow)[2];
      }
    v8f sc[2];
#pragma unroll
    for (int ct = 0; ct < 2; ++ct) {
#pragma unroll
      for (int e = 0; e < 8; ++e) sc[ct][e] = 0.f;
      sc[ct] = __builtin_amdgcn_wmma_f32_16x16x32_bf16(
          false, aq0.v, false, kb[ct][0].v, (short)0, sc[ct], false, false);
      sc[ct] = __builtin_amdgcn_wmma_f32_16x16x32_bf16(
          false, aq1.v, false, kb[ct][1].v, (short)0, sc[ct], false, false);
    }

    // ---- mask + online softmax (row stats per lane-half via shfl_xor) ----
    const int c0 = j0 + lane16, c1 = c0 + 16;
#pragma unroll
    for (int r = 0; r < 8; ++r) {
      const int rg = qbase + (hi ? 8 : 0) + r;
      float s0 = sc[0][r] * scale, s1 = sc[1][r] * scale;
      bool k0ok = ((c0 <= rg) && !((rg >= d) && (c0 < d))) ||
                  ((rg >= d) && (c0 == dm1));
      bool k1ok = ((c1 <= rg) && !((rg >= d) && (c1 < d))) ||
                  ((rg >= d) && (c1 == dm1));
      s0 = k0ok ? s0 : -1e30f;
      s1 = k1ok ? s1 : -1e30f;
      float rmax = fmaxf(s0, s1);
      for (int off = 1; off < 16; off <<= 1)
        rmax = fmaxf(rmax, __shfl_xor(rmax, off));
      const float nm = fmaxf(mrow[r], rmax);
      const float p0 = __expf(s0 - nm), p1 = __expf(s1 - nm);
      float rsum = p0 + p1;
      for (int off = 1; off < 16; off <<= 1) rsum += __shfl_xor(rsum, off);
      const float alpha = __expf(mrow[r] - nm);
      lrow[r] = lrow[r] * alpha + rsum;
      mrow[r] = nm;
#pragma unroll
      for (int nt = 0; nt < 4; ++nt) o[nt][r] *= alpha;
      const int prow = (hi ? 8 : 0) + r;
      p_lds[wv][prow * 32 + lane16]      = f2bf(p0);
      p_lds[wv][prow * 32 + lane16 + 16] = f2bf(p1);
    }

    // ---- re-read P as an A-fragment (16 rows x 32 keys) ----
    ABFrag pa;
    {
      const unsigned short* pp = &p_lds[wv][lane16 * 32 + (hi ? 8 : 0)];
      pa.u[0] = *(const uint4*)(pp);
      pa.u[1] = *(const uint4*)(pp + 16);
    }

    // ---- V tile must be resident now ----
    asm volatile("s_wait_asynccnt 0x0" ::: "memory");

    // ---- O += P * V : 4 output col-tiles over HD ----
#pragma unroll
    for (int nt = 0; nt < 4; ++nt) {
      ABFrag vb;
#pragma unroll
      for (int e = 0; e < 16; ++e)
        vb.s[e] = v_lds[wv][((hi ? 8 : 0) + (e < 8 ? e : e + 8)) * HD + nt * 16 + lane16];
      o[nt] = __builtin_amdgcn_wmma_f32_16x16x32_bf16(
          false, pa.v, false, vb.v, (short)0, o[nt], false, false);
    }
  }

  // ---- normalize and store (b, t, h*HD + hd) bf16 ----
#pragma unroll
  for (int nt = 0; nt < 4; ++nt)
#pragma unroll
    for (int r = 0; r < 8; ++r) {
      const int rg  = qbase + (hi ? 8 : 0) + r;
      const int col = h * HD + nt * 16 + lane16;
      attno[(size_t)(b * T + rg) * E + col] = f2bf(o[nt][r] / lrow[r]);
    }
}

// ---------------------------------------------------------------------------
// Host orchestration
// ---------------------------------------------------------------------------
extern "C" void kernel_launch(void* const* d_in, const int* in_sizes, int n_in,
                              void* d_out, int out_size, void* d_ws,
                              size_t ws_size, hipStream_t stream) {
  const float* wte     = (const float*)d_in[0];
  const float* wpe     = (const float*)d_in[1];
  const float* wpe_enc = (const float*)d_in[2];
  const float* ln1_g   = (const float*)d_in[3];
  const float* ln1_b   = (const float*)d_in[4];
  const float* w_in    = (const float*)d_in[5];
  const float* b_in    = (const float*)d_in[6];
  const float* w_out   = (const float*)d_in[7];
  const float* b_out   = (const float*)d_in[8];
  const float* ln2_g   = (const float*)d_in[9];
  const float* ln2_b   = (const float*)d_in[10];
  const float* w_fc    = (const float*)d_in[11];
  const float* b_fc    = (const float*)d_in[12];
  const float* w_proj  = (const float*)d_in[13];
  const float* b_proj  = (const float*)d_in[14];
  const float* lnf_g   = (const float*)d_in[15];
  const float* lnf_b   = (const float*)d_in[16];
  const float* w_head  = (const float*)d_in[17];
  const int*   tok     = (const int*)d_in[18];
  const int*   idxs    = (const int*)d_in[19];

  char* wsp = (char*)d_ws;
  auto alloc = [&](size_t bytes) {
    char* p = wsp;
    wsp += (bytes + 255) & ~(size_t)255;
    return p;
  };
  float*          x       = (float*)alloc((size_t)M * E * 4);
  unsigned short* hb      = (unsigned short*)alloc((size_t)M * E * 2);
  unsigned short* qkvb    = (unsigned short*)alloc((size_t)M * 3 * E * 2);
  unsigned short* attno   = (unsigned short*)alloc((size_t)M * E * 2);
  unsigned short* fcb     = (unsigned short*)alloc((size_t)M * 4 * E * 2);
  unsigned short* wb_in   = (unsigned short*)alloc((size_t)Lnum * 3 * E * E * 2);
  unsigned short* wb_out  = (unsigned short*)alloc((size_t)Lnum * E * E * 2);
  unsigned short* wb_fc   = (unsigned short*)alloc((size_t)Lnum * 4 * E * E * 2);
  unsigned short* wb_proj = (unsigned short*)alloc((size_t)Lnum * E * 4 * E * 2);
  unsigned short* wb_head = (unsigned short*)alloc((size_t)Vv * E * 2);

  auto cvt = [&](const float* s, unsigned short* dp, size_t n) {
    cvt_f32_bf16<<<dim3((unsigned)((n + 255) / 256)), dim3(256), 0, stream>>>(
        s, dp, (int)n);
  };
  cvt(w_in,   wb_in,   (size_t)Lnum * 3 * E * E);
  cvt(w_out,  wb_out,  (size_t)Lnum * E * E);
  cvt(w_fc,   wb_fc,   (size_t)Lnum * 4 * E * E);
  cvt(w_proj, wb_proj, (size_t)Lnum * E * 4 * E);
  cvt(w_head, wb_head, (size_t)Vv * E);

  embed_kernel<<<dim3(M), dim3(256), 0, stream>>>(wte, wpe, wpe_enc, tok, idxs, x);

  for (int i = 0; i < Lnum; ++i) {
    ln_kernel<<<dim3(M / 8), dim3(256), 0, stream>>>(x, ln1_g + i * E,
                                                     ln1_b + i * E, hb);
    // qkv = h @ w_in^T + b_in  -> bf16
    gemm_wmma<0, false, false, true, false>
        <<<dim3(3 * E / 256, M / 32), dim3(256), 0, stream>>>(
            hb, wb_in + (size_t)i * 3 * E * E, b_in + i * 3 * E, nullptr,
            nullptr, qkvb, nullptr, 3 * E, E);
    attn_wmma<<<dim3(T / 64, H, Bsz), dim3(128), 0, stream>>>(qkvb, idxs, attno);
    // x = x + o @ w_out^T + b_out
    gemm_wmma<0, true, true, false, false>
        <<<dim3(E / 256, M / 32), dim3(256), 0, stream>>>(
            attno, wb_out + (size_t)i * E * E, b_out + i * E, x, x, nullptr,
            nullptr, E, E);
    ln_kernel<<<dim3(M / 8), dim3(256), 0, stream>>>(x, ln2_g + i * E,
                                                     ln2_b + i * E, hb);
    // h2 = gelu(h @ w_fc^T + b_fc) -> bf16
    gemm_wmma<1, false, false, true, false>
        <<<dim3(4 * E / 256, M / 32), dim3(256), 0, stream>>>(
            hb, wb_fc + (size_t)i * 4 * E * E, b_fc + i * 4 * E, nullptr,
            nullptr, fcb, nullptr, 4 * E, E);
    // x = x + h2 @ w_proj^T + b_proj
    gemm_wmma<0, true, true, false, false>
        <<<dim3(E / 256, M / 32), dim3(256), 0, stream>>>(
            fcb, wb_proj + (size_t)i * E * 4 * E, b_proj + i * E, x, x, nullptr,
            nullptr, E, 4 * E);
  }

  ln_kernel<<<dim3(M / 8), dim3(256), 0, stream>>>(x, lnf_g, lnf_b, hb);
  // logits = xf @ w_head^T, NaN where t < idxs_dec[b]
  gemm_wmma<0, false, true, false, true>
      <<<dim3(Vv / 256, M / 32), dim3(256), 0, stream>>>(
          hb, wb_head, nullptr, nullptr, (float*)d_out, nullptr, idxs, Vv, E);
}
